// RPE_NET_36455682409151
// MI455X (gfx1250) — compile-verified, matrix operand order checked by reading, not run on hardware
//
#include <hip/hip_runtime.h>
#include <stdint.h>

// ---------------------------------------------------------------------------
// RPE-Net fused kernel for MI455X (gfx1250, wave32, WMMA bf16)
//   out = relu((feat[ni]-feat[pi]) @ W1^T + b1) + relu(emb[pid] @ W2^T + b2)
// P = 150000 paths, D_IN = D_OUT = 256.
//
// Roofline: 39.3 GFLOP vs 614 MB unique traffic (64 FLOP/B) -> memory bound
// only if matmul runs via bf16 WMMA. A-fragments are built in-register from
// the row gathers (each gathered row read 2x total), B-fragments stream from
// the 256 KB L2-resident bf16 weight copy, output is stored non-temporally.
// ---------------------------------------------------------------------------

typedef __attribute__((ext_vector_type(16))) __bf16 v16bf;
typedef __attribute__((ext_vector_type(8)))  float  v8f;

#define D_DIM 256

// 16x32 bf16 A-fragment (CDNA5 layout: lane l -> row l&15,
// K = 32*ks + (l>>4)*8 + {0..7, 16..23}) for the (child-parent) branch.
__device__ __forceinline__ v16bf diff_frag(const float* __restrict__ crow,
                                           const float* __restrict__ prow,
                                           int kbase) {
  const float4* c = (const float4*)(crow + kbase);
  const float4* p = (const float4*)(prow + kbase);
  float4 c0 = c[0], c1 = c[1], c2 = c[4], c3 = c[5];
  float4 p0 = p[0], p1 = p[1], p2 = p[4], p3 = p[5];
  v16bf r;
  r[0]  = (__bf16)(c0.x - p0.x); r[1]  = (__bf16)(c0.y - p0.y);
  r[2]  = (__bf16)(c0.z - p0.z); r[3]  = (__bf16)(c0.w - p0.w);
  r[4]  = (__bf16)(c1.x - p1.x); r[5]  = (__bf16)(c1.y - p1.y);
  r[6]  = (__bf16)(c1.z - p1.z); r[7]  = (__bf16)(c1.w - p1.w);
  r[8]  = (__bf16)(c2.x - p2.x); r[9]  = (__bf16)(c2.y - p2.y);
  r[10] = (__bf16)(c2.z - p2.z); r[11] = (__bf16)(c2.w - p2.w);
  r[12] = (__bf16)(c3.x - p3.x); r[13] = (__bf16)(c3.y - p3.y);
  r[14] = (__bf16)(c3.z - p3.z); r[15] = (__bf16)(c3.w - p3.w);
  return r;
}

// Same layout, single source row (curr_embed branch).
__device__ __forceinline__ v16bf row_frag(const float* __restrict__ erow,
                                          int kbase) {
  const float4* e = (const float4*)(erow + kbase);
  float4 e0 = e[0], e1 = e[1], e2 = e[4], e3 = e[5];
  v16bf r;
  r[0]  = (__bf16)e0.x; r[1]  = (__bf16)e0.y; r[2]  = (__bf16)e0.z; r[3]  = (__bf16)e0.w;
  r[4]  = (__bf16)e1.x; r[5]  = (__bf16)e1.y; r[6]  = (__bf16)e1.z; r[7]  = (__bf16)e1.w;
  r[8]  = (__bf16)e2.x; r[9]  = (__bf16)e2.y; r[10] = (__bf16)e2.z; r[11] = (__bf16)e2.w;
  r[12] = (__bf16)e3.x; r[13] = (__bf16)e3.y; r[14] = (__bf16)e3.z; r[15] = (__bf16)e3.w;
  return r;
}

// B-fragment: lane l holds output-column row W[ncol] of the bf16 weight copy
// (B = W^T so the B lane layout mirrors A with N in place of M).
__device__ __forceinline__ v16bf b_frag(const __bf16* __restrict__ wrow,
                                        int kbase) {
  union { uint4 u[2]; v16bf v; } f;
  f.u[0] = *(const uint4*)(wrow + kbase);        // K kbase..+7
  f.u[1] = *(const uint4*)(wrow + kbase + 16);   // K kbase+16..+23
  return f.v;
}

// ---------------------------------------------------------------------------
// One-time weight conversion f32 -> bf16 into workspace (256 KB, L2-resident).
// ---------------------------------------------------------------------------
__global__ void rpe_cvt_weights(const float* __restrict__ w1,
                                const float* __restrict__ w2,
                                __bf16* __restrict__ o1,
                                __bf16* __restrict__ o2, int n) {
  int i = blockIdx.x * blockDim.x + threadIdx.x;
  if (i < n) {
    o1[i] = (__bf16)w1[i];
    o2[i] = (__bf16)w2[i];
  }
}

// ---------------------------------------------------------------------------
// Main fused kernel: 256 threads = 8 waves, each wave owns a 16-row M-tile.
// N = 256 handled as 2 groups of 8 N-tiles (128 accumulator VGPRs live),
// so each gathered row is read (and converted) only twice.
// ---------------------------------------------------------------------------
__global__ __launch_bounds__(256)
void rpe_main(const float* __restrict__ features,
              const float* __restrict__ curr_embed,
              const float* __restrict__ b1,
              const float* __restrict__ b2,
              const __bf16* __restrict__ W1b,
              const __bf16* __restrict__ W2b,
              const int* __restrict__ path_id,
              const int* __restrict__ parent_idx,
              const int* __restrict__ neigh_idx,
              float* __restrict__ out, int P) {
  const int tid  = threadIdx.x;
  const int lane = tid & 31;
  const int wave = tid >> 5;
  const int rowbase = blockIdx.x * 128 + wave * 16;

  const int r     = lane & 15;        // M row within tile (A) / N col (B/D)
  const int mtop  = (lane >> 4) * 8;  // C/D: lanes 16-31 hold M = 8..15
  const int khalf = (lane >> 4) * 8;  // A/B: lanes 16-31 hold K offset +8

  // Per-lane gather indices, clamped so EXEC stays all-1s through the WMMAs.
  int prow  = rowbase + r;
  int prowc = prow < P ? prow : P - 1;
  const int pid = path_id[prowc];
  const int pi  = parent_idx[prowc];
  const int ni  = neigh_idx[prowc];
  const float* __restrict__ crow = features   + (size_t)ni  * D_DIM;
  const float* __restrict__ parw = features   + (size_t)pi  * D_DIM;
  const float* __restrict__ erow = curr_embed + (size_t)pid * D_DIM;

  const bool full_tile = (rowbase + 15) < P;

  for (int g = 0; g < 2; ++g) {
    v8f acc1[8], acc2[8];
    v8f z = {};
#pragma unroll
    for (int j = 0; j < 8; ++j) { acc1[j] = z; acc2[j] = z; }

    for (int ks = 0; ks < 8; ++ks) {         // K = 256 = 8 x 32
      const int kbase = ks * 32 + khalf;
      v16bf a1 = diff_frag(crow, parw, kbase);   // (child - parent) in bf16
      v16bf a2 = row_frag(erow, kbase);          // curr_embed row in bf16
#pragma unroll
      for (int j = 0; j < 8; ++j) {
        const int ncol = (g * 8 + j) * 16 + r;
        v16bf bw1 = b_frag(W1b + (size_t)ncol * D_DIM, kbase);
        v16bf bw2 = b_frag(W2b + (size_t)ncol * D_DIM, kbase);
        acc1[j] = __builtin_amdgcn_wmma_f32_16x16x32_bf16(
            false, a1, false, bw1, (short)0, acc1[j], false, false);
        acc2[j] = __builtin_amdgcn_wmma_f32_16x16x32_bf16(
            false, a2, false, bw2, (short)0, acc2[j], false, false);
      }
    }

    // Epilogue: bias + ReLU per branch, add, non-temporal store.
    // C/D layout: VGPR v, lane l -> M = 8*(l>>4)+v, N = l&15.
#pragma unroll
    for (int j = 0; j < 8; ++j) {
      const int ncol = (g * 8 + j) * 16 + r;
      const float bv1 = b1[ncol];
      const float bv2 = b2[ncol];
      if (full_tile) {
#pragma unroll
        for (int v = 0; v < 8; ++v) {
          const int orow = rowbase + mtop + v;
          float x1 = fmaxf(acc1[j][v] + bv1, 0.0f);
          float x2 = fmaxf(acc2[j][v] + bv2, 0.0f);
          __builtin_nontemporal_store(x1 + x2,
                                      out + (size_t)orow * D_DIM + ncol);
        }
      } else {
#pragma unroll
        for (int v = 0; v < 8; ++v) {
          const int orow = rowbase + mtop + v;
          if (orow < P) {
            float x1 = fmaxf(acc1[j][v] + bv1, 0.0f);
            float x2 = fmaxf(acc2[j][v] + bv2, 0.0f);
            __builtin_nontemporal_store(x1 + x2,
                                        out + (size_t)orow * D_DIM + ncol);
          }
        }
      }
    }
  }
}

// ---------------------------------------------------------------------------
// Launch: inputs in setup_inputs() order.
// ---------------------------------------------------------------------------
extern "C" void kernel_launch(void* const* d_in, const int* in_sizes, int n_in,
                              void* d_out, int out_size, void* d_ws, size_t ws_size,
                              hipStream_t stream) {
  const float* features   = (const float*)d_in[0];
  const float* curr_embed = (const float*)d_in[1];
  const float* W1         = (const float*)d_in[2];
  const float* b1         = (const float*)d_in[3];
  const float* W2         = (const float*)d_in[4];
  const float* b2         = (const float*)d_in[5];
  const int*   path_id    = (const int*)d_in[6];
  const int*   parent_idx = (const int*)d_in[7];
  const int*   neigh_idx  = (const int*)d_in[8];
  float* out = (float*)d_out;

  const int P  = in_sizes[6];          // 150000 paths
  const int nW = D_DIM * D_DIM;        // 65536 weights per matrix

  __bf16* W1b = (__bf16*)d_ws;
  __bf16* W2b = (__bf16*)((char*)d_ws + (size_t)nW * sizeof(__bf16));

  rpe_cvt_weights<<<(nW + 255) / 256, 256, 0, stream>>>(W1, W2, W1b, W2b, nW);

  const int blocks = (P + 127) / 128;  // 128 rows per 8-wave block
  rpe_main<<<blocks, 256, 0, stream>>>(features, curr_embed, b1, b2,
                                       W1b, W2b, path_id, parent_idx,
                                       neigh_idx, out, P);
}
